// Model_17291538333963
// MI455X (gfx1250) — matrix operand order repaired and checked by reading, not compiled, not run on hardware
//
#include <hip/hip_runtime.h>

typedef _Float16 half_t;
typedef __attribute__((ext_vector_type(16))) _Float16 v16h;
typedef __attribute__((ext_vector_type(8)))  float    v8f;

static constexpr int Bb = 16;      // batch
static constexpr int Tt = 96;      // T
static constexpr int Nn = 512;     // N == H
static constexpr int MT = 192;     // M*T
static constexpr int C4 = 768;     // 4*MT
static constexpr int BN = Bb * Nn; // 8192 rows of the "flattened" activations

__device__ __forceinline__ float lo_h(unsigned u) {
    unsigned short s = (unsigned short)(u & 0xffffu);
    return (float)__builtin_bit_cast(half_t, s);
}
__device__ __forceinline__ float hi_h(unsigned u) {
    unsigned short s = (unsigned short)(u >> 16);
    return (float)__builtin_bit_cast(half_t, s);
}

// ---------------------------------------------------------------------------
// Generic f32 -> f16 conversion
// ---------------------------------------------------------------------------
__global__ void k_f2h(const float* __restrict__ s, half_t* __restrict__ d, int n) {
    int i = blockIdx.x * blockDim.x + threadIdx.x;
    if (i < n) d[i] = (half_t)s[i];
}

// W_hh (1536,512) -> packed f16-pair layout Wp[k2][j] = {W[j][2k2], W[j][2k2+1]}
__global__ void k_whh_pack(const float* __restrict__ w, unsigned* __restrict__ wt) {
    int i = blockIdx.x * blockDim.x + threadIdx.x;
    if (i >= 256 * 1536) return;
    int k2 = i / 1536, j = i - k2 * 1536;
    half_t lo = (half_t)w[(size_t)j * 512 + 2 * k2];
    half_t hi = (half_t)w[(size_t)j * 512 + 2 * k2 + 1];
    unsigned ulo = (unsigned)__builtin_bit_cast(unsigned short, lo);
    unsigned uhi = (unsigned)__builtin_bit_cast(unsigned short, hi);
    wt[i] = ulo | (uhi << 16);
}

// Xall[m][b*96+t] = x[b][t][m]   (f16, 512 x 1536)
__global__ void k_xall(const float* __restrict__ x, half_t* __restrict__ d) {
    int i = blockIdx.x * blockDim.x + threadIdx.x;
    if (i >= Nn * Bb * Tt) return;
    int m = i / (Bb * Tt), c = i - m * (Bb * Tt);
    d[i] = (half_t)x[(size_t)c * Nn + m];
}

// ---------------------------------------------------------------------------
// GRU: input is zero, state identical across batch -> one sequential chain.
// One block of 512 threads; h lives in LDS; W_hh streamed as packed f16 pairs.
// ---------------------------------------------------------------------------
__global__ void __launch_bounds__(512) k_gru(const unsigned* __restrict__ Wp,
                                             const float* __restrict__ b_ih,
                                             const float* __restrict__ b_hh,
                                             float* __restrict__ outs) {
    __shared__ float h[512];
    int j = threadIdx.x;
    h[j] = 0.f;
    float bir = b_ih[j], biz = b_ih[j + 512], bin_ = b_ih[j + 1024];
    float bhr = b_hh[j], bhz = b_hh[j + 512], bhn = b_hh[j + 1024];
    __syncthreads();
    for (int n = 0; n < 512; ++n) {
        float dr = 0.f, dz = 0.f, dn = 0.f;
        for (int k2 = 0; k2 < 256; ++k2) {
            const unsigned* w = Wp + (size_t)k2 * 1536;
            unsigned wr = w[j], wz = w[j + 512], wn = w[j + 1024];
            float h0 = h[2 * k2], h1 = h[2 * k2 + 1];
            dr = fmaf(lo_h(wr), h0, fmaf(hi_h(wr), h1, dr));
            dz = fmaf(lo_h(wz), h0, fmaf(hi_h(wz), h1, dz));
            dn = fmaf(lo_h(wn), h0, fmaf(hi_h(wn), h1, dn));
        }
        float r  = 1.f / (1.f + expf(-(bir + bhr + dr)));
        float z  = 1.f / (1.f + expf(-(biz + bhz + dz)));
        float nn = tanhf(bin_ + r * (dn + bhn));
        float hn = (1.f - z) * nn + z * h[j];
        __syncthreads();
        h[j] = hn;
        outs[(size_t)n * 512 + j] = hn;
        __syncthreads();
    }
}

// key[i] = sum_n outs[n][i]*wk[n], query likewise
__global__ void __launch_bounds__(512) k_keyquery(const float* __restrict__ outs,
                                                  const float* __restrict__ wk,
                                                  const float* __restrict__ wq,
                                                  float* __restrict__ key,
                                                  float* __restrict__ query) {
    int i = threadIdx.x;
    float kq = 0.f, qq = 0.f;
    for (int n = 0; n < 512; ++n) {
        float o = outs[(size_t)n * 512 + i];
        kq = fmaf(o, wk[n], kq);
        qq = fmaf(o, wq[n], qq);
    }
    key[i] = kq; query[i] = qq;
}

// Row softmax of leaky(key[i]+query[j]); writes att and deg. 512 blocks x 256.
__global__ void __launch_bounds__(256) k_softmax(const float* __restrict__ key,
                                                 const float* __restrict__ query,
                                                 float* __restrict__ att,
                                                 float* __restrict__ deg) {
    __shared__ float red[256];
    int i = blockIdx.x, t = threadIdx.x;
    float ki = key[i];
    float e0 = ki + query[t], e1 = ki + query[t + 256];
    e0 = e0 > 0.f ? e0 : 0.2f * e0;
    e1 = e1 > 0.f ? e1 : 0.2f * e1;
    red[t] = fmaxf(e0, e1); __syncthreads();
    for (int s = 128; s > 0; s >>= 1) { if (t < s) red[t] = fmaxf(red[t], red[t + s]); __syncthreads(); }
    float m = red[0]; __syncthreads();
    float x0 = expf(e0 - m), x1 = expf(e1 - m);
    red[t] = x0 + x1; __syncthreads();
    for (int s = 128; s > 0; s >>= 1) { if (t < s) red[t] += red[t + s]; __syncthreads(); }
    float inv = 1.f / red[0]; __syncthreads();
    float a0 = x0 * inv, a1 = x1 * inv;
    att[(size_t)i * 512 + t] = a0;
    att[(size_t)i * 512 + t + 256] = a1;
    red[t] = a0 + a1; __syncthreads();
    for (int s = 128; s > 0; s >>= 1) { if (t < s) red[t] += red[t + s]; __syncthreads(); }
    if (t == 0) deg[i] = red[0];
}

// Normalized Laplacian (symmetric), written in f32 and f16
__global__ void k_lap(const float* __restrict__ att, const float* __restrict__ deg,
                      float* __restrict__ lap_f, half_t* __restrict__ lap_h) {
    int idx = blockIdx.x * blockDim.x + threadIdx.x;
    if (idx >= 512 * 512) return;
    int i = idx >> 9, j = idx & 511;
    float sym = 0.5f * (att[(size_t)i * 512 + j] + att[(size_t)j * 512 + i]);
    float di = 1.f / (sqrtf(deg[i]) + 1e-7f);
    float dj = 1.f / (sqrtf(deg[j]) + 1e-7f);
    float v = di * (((i == j) ? deg[i] : 0.f) - sym) * dj;
    lap_f[idx] = v;
    lap_h[idx] = (half_t)v;
}

// ---------------------------------------------------------------------------
// Tiled WMMA GEMM:  C = alpha * A @ op(B) [+ beta*Cadd] [+ bias] [act]
//   A: (M,K) f16 row-major, lda
//   TRANS_B=true : B = W (N,K) row-major (computes A@W^T), ldb=K
//   TRANS_B=false: B: (K,N) row-major, ldb=N
// Block: 256 threads (8 waves), tile 128x128, K-step 32.
// Interior tiles: GLOBAL_LOAD_ASYNC_TO_LDS_B128 with double-buffered LDS so
// the DMA of tile i+1 overlaps the WMMAs of tile i (ASYNCcnt, in-order).
// Edge tiles: zero-filled manual staging, chosen once per block.
// ---------------------------------------------------------------------------
static constexpr int TIL_M = 128, TIL_N = 128, TIL_K = 32, LSTR = 40;

__device__ __forceinline__ void async_copy_b128(const half_t* gp, half_t* lp) {
    unsigned lds_addr = (unsigned)(uintptr_t)lp;
    unsigned long long gaddr = (unsigned long long)(uintptr_t)gp;
    asm volatile("global_load_async_to_lds_b128 %0, %1, off"
                 :: "v"(lds_addr), "v"(gaddr) : "memory");
}
__device__ __forceinline__ void wait_async0() { asm volatile("s_wait_asynccnt 0x0" ::: "memory"); }
__device__ __forceinline__ void wait_async1() { asm volatile("s_wait_asynccnt 0x1" ::: "memory"); }
__device__ __forceinline__ void wait_async2() { asm volatile("s_wait_asynccnt 0x2" ::: "memory"); }

template<bool TRANS_B, bool HAS_BIAS, bool HAS_CADD, int ACT, bool OUT_HALF>
__global__ void __launch_bounds__(256) wmma_gemm(
    const half_t* __restrict__ A, const half_t* __restrict__ Bm,
    const float* __restrict__ bias, const float* __restrict__ Cadd,
    float alpha, float beta, void* __restrict__ Cout,
    int M, int N, int K, int lda, int ldb, int ldc)
{
    __shared__ half_t As[2][TIL_M * LSTR];
    __shared__ half_t Bs[2][TIL_N * LSTR];
    int tid = threadIdx.x;
    int wave = tid >> 5, lane = tid & 31;
    int wm = wave >> 2, wn = wave & 3;      // 2 waves in M, 4 in N
    int r = lane & 15, hs = lane >> 4;
    int m0 = blockIdx.y * TIL_M;
    int n0 = blockIdx.x * TIL_N;
    bool a_full = (m0 + TIL_M <= M);
    bool b_full = (n0 + TIL_N <= N);

    v8f acc[4][2];
#pragma unroll
    for (int a = 0; a < 4; ++a)
#pragma unroll
        for (int b = 0; b < 2; ++b)
#pragma unroll
            for (int e = 0; e < 8; ++e) acc[a][b][e] = 0.f;

    // per-thread staging coordinates
    int ar = tid >> 1, aseg = (tid & 1) * 16;         // A and NT-B: 2 thr/row
    const half_t* a_g = A + (size_t)(m0 + ar) * lda + aseg;
    const half_t* b_g = Bm + (size_t)(n0 + ar) * ldb + aseg;   // NT only
    int kk = tid >> 3, nseg = (tid & 7) * 16;                  // NN only

    auto compute_tile = [&](const half_t* Asb, const half_t* Bsb) {
        v16h af[4], bf[2];
#pragma unroll
        for (int mt = 0; mt < 4; ++mt) {
            const half_t* ap = &Asb[(wm * 64 + mt * 16 + r) * LSTR];
#pragma unroll
            for (int i = 0; i < 8; ++i) {
                int k = ((i >> 2) << 4) + (hs << 3) + ((i & 3) << 1);
                af[mt][2 * i] = ap[k];
                af[mt][2 * i + 1] = ap[k + 1];
            }
        }
#pragma unroll
        for (int nt = 0; nt < 2; ++nt) {
            const half_t* bp = &Bsb[(wn * 32 + nt * 16 + r) * LSTR];
#pragma unroll
            for (int i = 0; i < 8; ++i) {
                int k = ((i >> 2) << 4) + (hs << 3) + ((i & 3) << 1);
                bf[nt][2 * i] = bp[k];
                bf[nt][2 * i + 1] = bp[k + 1];
            }
        }
#pragma unroll
        for (int mt = 0; mt < 4; ++mt)
#pragma unroll
            for (int nt = 0; nt < 2; ++nt)
                acc[mt][nt] = __builtin_amdgcn_wmma_f32_16x16x32_f16(
                    false, af[mt], false, bf[nt], (short)0, acc[mt][nt], false, false);
    };

    auto stage_b_trans = [&](int k0, half_t* Bsb) {  // (K,N) tile -> [n][k] LDS
        const half_t* src = Bm + (size_t)(k0 + kk) * ldb + n0 + nseg;
        if (n0 + nseg + 16 <= N) {
            uint4 bv = *(const uint4*)src;
            const half_t* tv = (const half_t*)&bv;
#pragma unroll
            for (int e = 0; e < 16; ++e) Bsb[(nseg + e) * LSTR + kk] = tv[e];
        } else {
#pragma unroll
            for (int e = 0; e < 16; ++e) {
                int gn2 = n0 + nseg + e;
                Bsb[(nseg + e) * LSTR + kk] = (gn2 < N) ? src[e] : (half_t)0.f;
            }
        }
    };

    int nk = K / TIL_K;  // all K in this app are multiples of 32
    bool fast = TRANS_B ? (a_full && b_full) : a_full;

    if (fast) {
        // --- double-buffered async pipeline ---
        async_copy_b128(a_g, &As[0][ar * LSTR + aseg]);
        if constexpr (TRANS_B) async_copy_b128(b_g, &Bs[0][ar * LSTR + aseg]);
        for (int i = 0; i < nk; ++i) {
            int cur = i & 1;
            bool has_next = (i + 1 < nk);
            if (has_next) {
                int nxt = cur ^ 1;
                async_copy_b128(a_g + (size_t)(i + 1) * TIL_K, &As[nxt][ar * LSTR + aseg]);
                if constexpr (TRANS_B)
                    async_copy_b128(b_g + (size_t)(i + 1) * TIL_K, &Bs[nxt][ar * LSTR + aseg]);
            }
            if constexpr (!TRANS_B) stage_b_trans(i * TIL_K, &Bs[cur][0]);
            if (has_next) { if constexpr (TRANS_B) wait_async2(); else wait_async1(); }
            else wait_async0();
            __syncthreads();
            compute_tile(&As[cur][0], &Bs[cur][0]);
            __syncthreads();
        }
    } else {
        // --- edge path: zero-filled manual staging, single buffer ---
        for (int i = 0; i < nk; ++i) {
            int k0 = i * TIL_K;
            {
                int grow = m0 + ar;
                uint4 av = {0u, 0u, 0u, 0u};
                if (grow < M) av = *(const uint4*)(a_g + (size_t)k0);
                *(uint4*)(&As[0][ar * LSTR + aseg]) = av;
            }
            if constexpr (TRANS_B) {
                int gn = n0 + ar;
                uint4 bv = {0u, 0u, 0u, 0u};
                if (gn < N) bv = *(const uint4*)(b_g + (size_t)k0);
                *(uint4*)(&Bs[0][ar * LSTR + aseg]) = bv;
            } else {
                stage_b_trans(k0, &Bs[0][0]);
            }
            __syncthreads();
            compute_tile(&As[0][0], &Bs[0][0]);
            __syncthreads();
        }
    }

    // Epilogue + store. C/D layout: lane r -> column, vgpr v + hs*8 -> row.
#pragma unroll
    for (int mt = 0; mt < 4; ++mt)
#pragma unroll
        for (int nt = 0; nt < 2; ++nt) {
            int col = n0 + wn * 32 + nt * 16 + r;
#pragma unroll
            for (int v = 0; v < 8; ++v) {
                int row = m0 + wm * 64 + mt * 16 + hs * 8 + v;
                if (row < M && col < N) {
                    float val = alpha * acc[mt][nt][v];
                    if constexpr (HAS_CADD) val += beta * Cadd[(size_t)row * ldc + col];
                    if constexpr (HAS_BIAS) val += bias[col];
                    if constexpr (ACT == 1) val = 1.f / (1.f + expf(-val));
                    else if constexpr (ACT == 2) val = val > 0.f ? val : 0.01f * val;
                    if constexpr (OUT_HALF) ((half_t*)Cout)[(size_t)row * ldc + col] = (half_t)val;
                    else                    ((float*)Cout)[(size_t)row * ldc + col] = val;
                }
            }
        }
}

// ---------------------------------------------------------------------------
// Length-4 FFT of [0, g1, g2, g3] -> GLU-A inputs (f16)
// ---------------------------------------------------------------------------
__global__ void k_fftpack(const float* __restrict__ G1, const float* __restrict__ G2,
                          const float* __restrict__ G3,
                          half_t* __restrict__ re, half_t* __restrict__ im) {
    int idx = blockIdx.x * blockDim.x + threadIdx.x;
    if (idx >= BN * 4 * Tt) return;
    int row = idx / (4 * Tt), f = idx - row * (4 * Tt);
    int kf = f / Tt, t = f - kf * Tt;
    int b = row >> 9, n = row & 511;
    size_t c = (size_t)n * (Bb * Tt) + b * Tt + t;
    float g1 = G1[c], g2 = G2[c], g3 = G3[c];
    float rr, ii;
    if (kf == 0)      { rr = g1 + g2 + g3;   ii = 0.f; }
    else if (kf == 1) { rr = -g2;            ii = g3 - g1; }
    else if (kf == 2) { rr = -g1 + g2 - g3;  ii = 0.f; }
    else              { rr = -g2;            ii = g1 - g3; }
    re[idx] = (half_t)rr;
    im[idx] = (half_t)ii;
}

// GLU gate combine: dst = L * sigmoid(R)
template<bool OUT_HALF>
__global__ void k_glu_combine(const float* __restrict__ L, const float* __restrict__ R,
                              void* __restrict__ dst, int n) {
    int i = blockIdx.x * blockDim.x + threadIdx.x;
    if (i >= n) return;
    float v = L[i] * (1.f / (1.f + expf(-R[i])));
    if constexpr (OUT_HALF) ((half_t*)dst)[i] = (half_t)v;
    else                    ((float*)dst)[i] = v;
}

// irfft(n=4) over freq axis, packed (k,t)-major for the einsum GEMM (f16 out)
__global__ void k_irfft(const float* __restrict__ re, const float* __restrict__ im,
                        half_t* __restrict__ g) {
    int idx = blockIdx.x * blockDim.x + threadIdx.x;
    if (idx >= BN * C4) return;
    int row = idx / C4, f = idx - row * C4;
    int k = f / MT, u = f - k * MT;
    size_t base = (size_t)row * C4 + u;
    float R0 = re[base], R1 = re[base + MT], R2 = re[base + 2 * MT];
    float I1 = im[base + MT];
    float v;
    if (k == 0)      v = R0 + 2.f * R1 + R2;
    else if (k == 1) v = R0 - 2.f * I1 - R2;
    else if (k == 2) v = R0 - 2.f * R1 + R2;
    else             v = R0 + 2.f * I1 - R2;
    g[idx] = (half_t)(0.25f * v);
}

// Final matvec: out[row] = h1[row,:] . fc2_W + fc2_b
__global__ void k_fc2(const float* __restrict__ h1, const float* __restrict__ w,
                      const float* __restrict__ b, float* __restrict__ out) {
    int row = blockIdx.x * blockDim.x + threadIdx.x;
    if (row >= BN) return;
    float s = b[0];
    const float* hp = h1 + (size_t)row * Tt;
    for (int k = 0; k < Tt; ++k) s = fmaf(hp[k], w[k], s);
    out[row] = s;
}

// ---------------------------------------------------------------------------
extern "C" void kernel_launch(void* const* d_in, const int* in_sizes, int n_in,
                              void* d_out, int out_size, void* d_ws, size_t ws_size,
                              hipStream_t stream) {
    const float* x       = (const float*)d_in[0];
    const float* W_hh    = (const float*)d_in[2];
    const float* b_ih    = (const float*)d_in[3];
    const float* b_hh    = (const float*)d_in[4];
    const float* wk      = (const float*)d_in[5];
    const float* wq      = (const float*)d_in[6];
    const float* wparam  = (const float*)d_in[7];
    const float* fl_W    = (const float*)d_in[8];
    const float* fl_b    = (const float*)d_in[9];
    const float* frl_W   = (const float*)d_in[10];
    const float* frl_b   = (const float*)d_in[11];
    const float* gluA_lW = (const float*)d_in[12];
    const float* gluA_lb = (const float*)d_in[13];
    const float* gluA_rW = (const float*)d_in[14];
    const float* gluA_rb = (const float*)d_in[15];
    const float* gluB_lW = (const float*)d_in[16];
    const float* gluB_lb = (const float*)d_in[17];
    const float* gluB_rW = (const float*)d_in[18];
    const float* gluB_rb = (const float*)d_in[19];
    const float* fc1_W   = (const float*)d_in[20];
    const float* fc1_b   = (const float*)d_in[21];
    const float* fc2_W   = (const float*)d_in[22];
    const float* fc2_b   = (const float*)d_in[23];

    float* forecast = (float*)d_out;            // (B,1,N) flat = 8192
    float* att      = (float*)d_out + BN;       // (512,512)

    char* ws = (char*)d_ws;
    size_t off = 0;
    auto alloc = [&](size_t bytes) -> void* {
        void* p = ws + off;
        off = (off + bytes + 255) & ~(size_t)255;
        return p;
    };

    unsigned* Whh_p = (unsigned*)alloc((size_t)256 * 1536 * 4);
    float*  outs   = (float*)alloc(512u * 512u * 4u);
    float*  key    = (float*)alloc(512u * 4u);
    float*  query  = (float*)alloc(512u * 4u);
    float*  deg    = (float*)alloc(512u * 4u);
    float*  lap_f  = (float*)alloc(512u * 512u * 4u);
    half_t* lap_h  = (half_t*)alloc(512u * 512u * 2u);
    float*  L3_f   = (float*)alloc(512u * 512u * 4u);
    half_t* L3_h   = (half_t*)alloc(512u * 512u * 2u);
    float*  L4_f   = (float*)alloc(512u * 512u * 4u);
    half_t* L4_h   = (half_t*)alloc(512u * 512u * 2u);
    half_t* Xall   = (half_t*)alloc((size_t)512 * 1536 * 2);
    float*  G1     = (float*)alloc((size_t)512 * 1536 * 4);
    float*  G2     = (float*)alloc((size_t)512 * 1536 * 4);
    float*  G3     = (float*)alloc((size_t)512 * 1536 * 4);
    half_t* reA    = (half_t*)alloc((size_t)BN * 384 * 2);
    half_t* imA    = (half_t*)alloc((size_t)BN * 384 * 2);
    half_t* reC    = (half_t*)alloc((size_t)BN * C4 * 2);
    half_t* imC    = (half_t*)alloc((size_t)BN * C4 * 2);
    float*  Lr     = (float*)alloc((size_t)BN * C4 * 4);
    float*  Rr     = (float*)alloc((size_t)BN * C4 * 4);
    float*  Li     = (float*)alloc((size_t)BN * C4 * 4);
    float*  Ri     = (float*)alloc((size_t)BN * C4 * 4);
    half_t* Gflat  = (half_t*)alloc((size_t)BN * C4 * 2);
    half_t* igft_h = (half_t*)alloc((size_t)BN * MT * 2);
    half_t* src_h  = (half_t*)alloc((size_t)BN * MT * 2);
    half_t* fo_h   = (half_t*)alloc((size_t)BN * Tt * 2);
    float*  h1_f   = (float*)alloc((size_t)BN * Tt * 4);
    half_t* wAl    = (half_t*)alloc((size_t)2 * C4 * 384 * 2);
    half_t* wAr    = (half_t*)alloc((size_t)2 * C4 * 384 * 2);
    half_t* wBl    = (half_t*)alloc((size_t)4 * C4 * C4 * 2);
    half_t* wBr    = (half_t*)alloc((size_t)4 * C4 * C4 * 2);
    half_t* wFl    = (half_t*)alloc((size_t)MT * MT * 2);
    half_t* wBig   = (half_t*)alloc((size_t)C4 * MT * 2);
    half_t* wFrl   = (half_t*)alloc((size_t)Tt * MT * 2);
    half_t* wFc1   = (half_t*)alloc((size_t)Tt * Tt * 2);

    auto cdiv = [](int a, int b) { return (a + b - 1) / b; };
    auto f2h = [&](const float* s, half_t* d, int n) {
        k_f2h<<<cdiv(n, 256), 256, 0, stream>>>(s, d, n);
    };

    // --- weight conversions (f32 -> f16) ---
    f2h(gluA_lW, wAl, 2 * C4 * 384);
    f2h(gluA_rW, wAr, 2 * C4 * 384);
    f2h(gluB_lW, wBl, 4 * C4 * C4);
    f2h(gluB_rW, wBr, 4 * C4 * C4);
    f2h(fl_W,    wFl, MT * MT);
    f2h(wparam,  wBig, 4 * MT * MT);   // (4,192,192) viewed as (768,192) row-major
    f2h(frl_W,   wFrl, Tt * MT);
    f2h(fc1_W,   wFc1, Tt * Tt);
    k_whh_pack<<<cdiv(256 * 1536, 256), 256, 0, stream>>>(W_hh, Whh_p);
    k_xall<<<cdiv(512 * 1536, 256), 256, 0, stream>>>(x, Xall);

    // --- GRU chain + attention + Laplacian ---
    k_gru<<<1, 512, 0, stream>>>(Whh_p, b_ih, b_hh, outs);
    k_keyquery<<<1, 512, 0, stream>>>(outs, wk, wq, key, query);
    k_softmax<<<512, 256, 0, stream>>>(key, query, att, deg);
    k_lap<<<cdiv(512 * 512, 256), 256, 0, stream>>>(att, deg, lap_f, lap_h);

#define GEMM(TB, HB, HC, ACT, OH, Ap, Bp, biasp, caddp, alpha, beta, Cp, M, N, K, lda, ldb, ldc) \
    wmma_gemm<TB, HB, HC, ACT, OH><<<dim3(cdiv(N, 128), cdiv(M, 128)), 256, 0, stream>>>( \
        Ap, Bp, biasp, caddp, alpha, beta, Cp, M, N, K, lda, ldb, ldc)

    // --- Chebyshev: L3 = 2*lap@lap ; L4 = 2*lap@L3 - lap  (lap, L3 symmetric) ---
    GEMM(true, false, false, 0, false, lap_h, lap_h, (const float*)nullptr, (const float*)nullptr,
         2.f, 0.f, L3_f, 512, 512, 512, 512, 512, 512);
    f2h(L3_f, L3_h, 512 * 512);
    GEMM(true, false, true, 0, false, lap_h, L3_h, (const float*)nullptr, lap_f,
         2.f, -1.f, L4_f, 512, 512, 512, 512, 512, 512);
    f2h(L4_f, L4_h, 512 * 512);

    // --- gft: Gk = chebk @ Xall  (NN, 512 x 1536 x 512) ---
    GEMM(false, false, false, 0, false, lap_h, Xall, (const float*)nullptr, (const float*)nullptr,
         1.f, 0.f, G1, 512, 1536, 512, 512, 1536, 1536);
    GEMM(false, false, false, 0, false, L3_h, Xall, (const float*)nullptr, (const float*)nullptr,
         1.f, 0.f, G2, 512, 1536, 512, 512, 1536, 1536);
    GEMM(false, false, false, 0, false, L4_h, Xall, (const float*)nullptr, (const float*)nullptr,
         1.f, 0.f, G3, 512, 1536, 512, 512, 1536, 1536);

    // --- FFT(4) -> real/imag (8192 x 384, f16) ---
    k_fftpack<<<cdiv(BN * 384, 256), 256, 0, stream>>>(G1, G2, G3, reA, imA);

    // --- GLU A ---
    GEMM(true, true, false, 0, false, reA, wAl, gluA_lb, (const float*)nullptr,
         1.f, 0.f, Lr, BN, C4, 384, 384, 384, C4);
    GEMM(true, true, false, 0, false, reA, wAr, gluA_rb, (const float*)nullptr,
         1.f, 0.f, Rr, BN, C4, 384, 384, 384, C4);
    k_glu_combine<true><<<cdiv(BN * C4, 256), 256, 0, stream>>>(Lr, Rr, reC, BN * C4);
    GEMM(true, true, false, 0, false, imA, wAl + (size_t)C4 * 384, gluA_lb + C4, (const float*)nullptr,
         1.f, 0.f, Li, BN, C4, 384, 384, 384, C4);
    GEMM(true, true, false, 0, false, imA, wAr + (size_t)C4 * 384, gluA_rb + C4, (const float*)nullptr,
         1.f, 0.f, Ri, BN, C4, 384, 384, 384, C4);
    k_glu_combine<true><<<cdiv(BN * C4, 256), 256, 0, stream>>>(Li, Ri, imC, BN * C4);

    // --- GLU B x2 (real uses indices 0,2; imag uses 1,3). Final combine -> f32 in-place ---
    for (int it = 0; it < 2; ++it) {
        int iR = 2 * it, iI = 2 * it + 1;
        GEMM(true, true, false, 0, false, reC, wBl + (size_t)iR * C4 * C4, gluB_lb + iR * C4,
             (const float*)nullptr, 1.f, 0.f, Lr, BN, C4, C4, C4, C4, C4);
        GEMM(true, true, false, 0, false, reC, wBr + (size_t)iR * C4 * C4, gluB_rb + iR * C4,
             (const float*)nullptr, 1.f, 0.f, Rr, BN, C4, C4, C4, C4, C4);
        GEMM(true, true, false, 0, false, imC, wBl + (size_t)iI * C4 * C4, gluB_lb + iI * C4,
             (const float*)nullptr, 1.f, 0.f, Li, BN, C4, C4, C4, C4, C4);
        GEMM(true, true, false, 0, false, imC, wBr + (size_t)iI * C4 * C4, gluB_rb + iI * C4,
             (const float*)nullptr, 1.f, 0.f, Ri, BN, C4, C4, C4, C4, C4);
        if (it == 0) {
            k_glu_combine<true><<<cdiv(BN * C4, 256), 256, 0, stream>>>(Lr, Rr, reC, BN * C4);
            k_glu_combine<true><<<cdiv(BN * C4, 256), 256, 0, stream>>>(Li, Ri, imC, BN * C4);
        } else {
            k_glu_combine<false><<<cdiv(BN * C4, 256), 256, 0, stream>>>(Lr, Rr, Lr, BN * C4);
            k_glu_combine<false><<<cdiv(BN * C4, 256), 256, 0, stream>>>(Li, Ri, Li, BN * C4);
        }
    }

    // --- irfft(4) + pack (k,t)-major -> Gflat (8192 x 768, f16) ---
    k_irfft<<<cdiv(BN * C4, 256), 256, 0, stream>>>(Lr, Li, Gflat);

    // --- igft = Gflat @ weight_param_flat (NN, 8192 x 192 x 768) -> f16 ---
    GEMM(false, false, false, 0, true, Gflat, wBig, (const float*)nullptr, (const float*)nullptr,
         1.f, 0.f, igft_h, BN, MT, C4, C4, MT, MT);
    // --- src = sigmoid(igft @ fl_W^T + fl_b) -> f16 ---
    GEMM(true, true, false, 1, true, igft_h, wFl, fl_b, (const float*)nullptr,
         1.f, 0.f, src_h, BN, MT, MT, MT, MT, MT);
    // --- fo = src @ frl_W^T + frl_b -> f16 ---
    GEMM(true, true, false, 0, true, src_h, wFrl, frl_b, (const float*)nullptr,
         1.f, 0.f, fo_h, BN, Tt, MT, MT, MT, Tt);
    // --- h1 = leaky(fo @ fc1_W^T + fc1_b) -> f32 ---
    GEMM(true, true, false, 2, false, fo_h, wFc1, fc1_b, (const float*)nullptr,
         1.f, 0.f, h1_f, BN, Tt, Tt, Tt, Tt, Tt);
    // --- out = h1 @ fc2_W^T + fc2_b -> forecast ---
    k_fc2<<<cdiv(BN, 256), 256, 0, stream>>>(h1_f, fc2_W, fc2_b, forecast);

#undef GEMM
    (void)in_sizes; (void)n_in; (void)out_size; (void)ws_size;
}
